// GATEncoder_89635967467600
// MI455X (gfx1250) — compile-verified, matrix operand order checked by reading, not run on hardware
//
#include <hip/hip_runtime.h>
#include <hip/hip_bf16.h>

typedef __attribute__((ext_vector_type(16))) __bf16 v16bf;
typedef __attribute__((ext_vector_type(8)))  float  v8f;

#define GAT_IN   512
#define GAT_H    4
#define GAT_C1   256
#define GAT_F1   1024   /* H * C1 */
#define GAT_OUT  128
#define NEG_SLOPE 0.2f
#define LDK 40          /* padded LDS row stride in bf16 elems (80B, 16B-aligned) */

// ---------------- helpers ----------------
__device__ __forceinline__ unsigned short f2bf(float f) {
  unsigned u = __float_as_uint(f);
  u += 0x7FFFu + ((u >> 16) & 1u);          // round to nearest even
  return (unsigned short)(u >> 16);
}
// order-preserving float<->uint for atomicMax-based segment max
__device__ __forceinline__ unsigned encf(float f) {
  unsigned u = __float_as_uint(f);
  return (u & 0x80000000u) ? ~u : (u | 0x80000000u);
}
__device__ __forceinline__ float decf(unsigned e) {
  return __uint_as_float((e & 0x80000000u) ? (e ^ 0x80000000u) : ~e);
}

// ---------------- utility kernels ----------------
__global__ __launch_bounds__(256) void k_f32_to_bf16(const float* __restrict__ in,
                                                     unsigned short* __restrict__ out, int n) {
  int i = blockIdx.x * blockDim.x + threadIdx.x;
  if (i < n) out[i] = f2bf(in[i]);
}

__global__ __launch_bounds__(256) void k_zero_u32(unsigned* __restrict__ p, int n) {
  int i = blockIdx.x * blockDim.x + threadIdx.x;
  if (i < n) p[i] = 0u;
}

__global__ __launch_bounds__(256) void k_init_bias(float* __restrict__ out,
                                                   const float* __restrict__ b, int n, int c) {
  int i = blockIdx.x * blockDim.x + threadIdx.x;
  if (i < n) out[i] = b[i % c];
}

__global__ __launch_bounds__(256) void k_bias_relu_bf16(const float* __restrict__ in,
                                                        const float* __restrict__ b,
                                                        unsigned short* __restrict__ outbf,
                                                        int n, int F) {
  int i = blockIdx.x * blockDim.x + threadIdx.x;
  if (i >= n) return;
  float v = in[i] + b[i % F];
  v = v > 0.f ? v : 0.f;
  outbf[i] = f2bf(v);
}

// ---------------- bf16 WMMA GEMM: C[M,N] f32 = A[M,K] bf16 @ B[K,N] bf16 ----------------
// 64x64 block tile, 4 waves (2x2), each wave does 2x2 16x16 WMMA tiles, K-step 32.
__global__ __launch_bounds__(128) void k_gemm_bf16(const unsigned short* __restrict__ A,
                                                   const unsigned short* __restrict__ B,
                                                   float* __restrict__ C,
                                                   int M, int N, int K) {
  __shared__ unsigned short lA[64 * LDK];   // [row][k]
  __shared__ unsigned short lB[64 * LDK];   // [n][k]  (B transposed)
  const int bm = blockIdx.x * 64;
  const int bn = blockIdx.y * 64;
  const int t = threadIdx.x;
  const int wave = t >> 5, lane = t & 31;
  const int wr = wave >> 1, wc = wave & 1;
  const int r = lane & 15, hh = lane >> 4;

  v8f acc[2][2] = {};

  for (int k0 = 0; k0 < K; k0 += 32) {
    // stage A tile: 64 rows x 32 bf16, 16B chunks
#pragma unroll
    for (int i = 0; i < 2; ++i) {
      int chunk = t + i * 128;              // 0..255
      int row = chunk >> 2;
      int kc = (chunk & 3) * 8;             // bf16 offset within row
      uint4 v = make_uint4(0u, 0u, 0u, 0u);
      if (bm + row < M)
        v = *(const uint4*)(A + (size_t)(bm + row) * K + k0 + kc);
      *(uint4*)(lA + row * LDK + kc) = v;
    }
    // stage B tile transposed: lB[n][k] = B[k0+k][bn+n]
    for (int idx = t; idx < 2048; idx += 128) {
      int k = idx >> 6;
      int n = idx & 63;
      lB[n * LDK + k] = B[(size_t)(k0 + k) * N + bn + n];
    }
    __syncthreads();

    union Frag { uint4 q[2]; v16bf v; };
    Frag af[2], bfr[2];
#pragma unroll
    for (int mt = 0; mt < 2; ++mt) {
      const uint4* p = (const uint4*)(lA + (wr * 32 + mt * 16 + r) * LDK);
      af[mt].q[0] = p[hh];        // K = {0..7}  + 8*hh
      af[mt].q[1] = p[2 + hh];    // K = {16..23}+ 8*hh
    }
#pragma unroll
    for (int nt = 0; nt < 2; ++nt) {
      const uint4* p = (const uint4*)(lB + (wc * 32 + nt * 16 + r) * LDK);
      bfr[nt].q[0] = p[hh];
      bfr[nt].q[1] = p[2 + hh];
    }
#pragma unroll
    for (int mt = 0; mt < 2; ++mt)
#pragma unroll
      for (int nt = 0; nt < 2; ++nt)
        acc[mt][nt] = __builtin_amdgcn_wmma_f32_16x16x32_bf16(
            false, af[mt].v, false, bfr[nt].v, (short)0, acc[mt][nt], false, false);
    __syncthreads();
  }

  // epilogue: VGPR i / lane layout -> (m, n)
#pragma unroll
  for (int mt = 0; mt < 2; ++mt) {
#pragma unroll
    for (int nt = 0; nt < 2; ++nt) {
      int n = bn + wc * 32 + nt * 16 + r;
#pragma unroll
      for (int i = 0; i < 8; ++i) {
        int m = bm + wr * 32 + mt * 16 + hh * 8 + i;
        if (m < M) C[(size_t)m * N + n] = acc[mt][nt][i];
      }
    }
  }
}

// ---------------- edge kernels (one wave per (edge, head)) ----------------
__global__ __launch_bounds__(256) void k_edge_score(const float* __restrict__ xl,
                                                    const float* __restrict__ xr,
                                                    const float* __restrict__ att,
                                                    const int* __restrict__ src,
                                                    const int* __restrict__ dst,
                                                    float* __restrict__ s,
                                                    unsigned* __restrict__ smax,
                                                    int E, int H, int C) {
  int task = blockIdx.x * 8 + (threadIdx.x >> 5);
  if (task >= E * H) return;
  int lane = threadIdx.x & 31;
  int e = task / H, h = task - e * H;
  int si = src[e], di = dst[e];
  const float* pl = xl + (size_t)si * H * C + h * C;
  const float* pr = xr + (size_t)di * H * C + h * C;
  const float* pa = att + h * C;
  float acc = 0.f;
  for (int c = lane; c < C; c += 32) {
    float v = pl[c] + pr[c];
    v = v > 0.f ? v : NEG_SLOPE * v;
    acc += v * pa[c];
  }
#pragma unroll
  for (int off = 16; off > 0; off >>= 1)
    acc += __shfl_xor(acc, off, 32);
  if (lane == 0) {
    s[(size_t)e * H + h] = acc;
    atomicMax(&smax[di * H + h], encf(acc));
  }
}

__global__ __launch_bounds__(256) void k_edge_exp(const int* __restrict__ dst,
                                                  float* __restrict__ s,
                                                  const unsigned* __restrict__ smax,
                                                  float* __restrict__ denom, int E, int H) {
  int i = blockIdx.x * blockDim.x + threadIdx.x;
  if (i >= E * H) return;
  int e = i / H, h = i - e * H;
  int di = dst[e];
  float ex = __expf(s[i] - decf(smax[di * H + h]));
  s[i] = ex;
  atomicAdd(&denom[di * H + h], ex);
}

__global__ __launch_bounds__(256) void k_edge_scatter(const float* __restrict__ xl,
                                                      const int* __restrict__ src,
                                                      const int* __restrict__ dst,
                                                      const float* __restrict__ s,
                                                      const float* __restrict__ denom,
                                                      float* __restrict__ out,
                                                      int E, int H, int C) {
  int task = blockIdx.x * 8 + (threadIdx.x >> 5);
  if (task >= E * H) return;
  int lane = threadIdx.x & 31;
  int e = task / H, h = task - e * H;
  int si = src[e], di = dst[e];
  float alpha = s[(size_t)e * H + h] / denom[di * H + h];
  const float* pl = xl + (size_t)si * H * C + h * C;
  float* po = out + (size_t)di * H * C + h * C;
  for (int c = lane; c < C; c += 32)
    atomicAdd(&po[c], alpha * pl[c]);
}

// ---------------- launch ----------------
static inline int nblk(long long n, int b) { return (int)((n + b - 1) / b); }

extern "C" void kernel_launch(void* const* d_in, const int* in_sizes, int n_in,
                              void* d_out, int out_size, void* d_ws, size_t ws_size,
                              hipStream_t stream) {
  const float* x    = (const float*)d_in[0];
  const int*   ei   = (const int*)d_in[1];
  const float* Wl1  = (const float*)d_in[2];
  const float* Wr1  = (const float*)d_in[3];
  const float* att1 = (const float*)d_in[4];
  const float* b1   = (const float*)d_in[5];
  const float* Wl2  = (const float*)d_in[6];
  const float* Wr2  = (const float*)d_in[7];
  const float* att2 = (const float*)d_in[8];
  const float* b2   = (const float*)d_in[9];
  float* out = (float*)d_out;

  const int Nn = in_sizes[0] / GAT_IN;   // 10000
  const int E  = in_sizes[1] / 2;        // 320000
  const int* src = ei;
  const int* dst = ei + E;

  // workspace carve-up (256B aligned)
  char* ws = (char*)d_ws;
  size_t off = 0;
  auto alloc = [&](size_t bytes) -> void* {
    void* p = ws + off;
    off += (bytes + 255) & ~(size_t)255;
    return p;
  };
  unsigned short* x_bf  = (unsigned short*)alloc((size_t)Nn * GAT_IN * 2);
  unsigned short* w1l   = (unsigned short*)alloc((size_t)GAT_IN * GAT_F1 * 2);
  unsigned short* w1r   = (unsigned short*)alloc((size_t)GAT_IN * GAT_F1 * 2);
  float*          xl1   = (float*)alloc((size_t)Nn * GAT_F1 * 4);
  float*          xr1   = (float*)alloc((size_t)Nn * GAT_F1 * 4);
  float*          s1    = (float*)alloc((size_t)E * GAT_H * 4);
  unsigned*       smax1 = (unsigned*)alloc((size_t)Nn * GAT_H * 4);
  float*          den1  = (float*)alloc((size_t)Nn * GAT_H * 4);
  float*          out1  = (float*)alloc((size_t)Nn * GAT_F1 * 4);
  unsigned short* h_bf  = (unsigned short*)alloc((size_t)Nn * GAT_F1 * 2);
  unsigned short* w2l   = (unsigned short*)alloc((size_t)GAT_F1 * GAT_OUT * 2);
  unsigned short* w2r   = (unsigned short*)alloc((size_t)GAT_F1 * GAT_OUT * 2);
  float*          hl2   = (float*)alloc((size_t)Nn * GAT_OUT * 4);
  float*          hr2   = (float*)alloc((size_t)Nn * GAT_OUT * 4);
  float*          s2    = (float*)alloc((size_t)E * 4);
  unsigned*       smax2 = (unsigned*)alloc((size_t)Nn * 4);
  float*          den2  = (float*)alloc((size_t)Nn * 4);
  (void)ws_size; (void)n_in; (void)out_size;

  // ---- conv1 projections (bf16 WMMA GEMMs) ----
  k_f32_to_bf16<<<nblk((long long)Nn * GAT_IN, 256), 256, 0, stream>>>(x, x_bf, Nn * GAT_IN);
  k_f32_to_bf16<<<nblk((long long)GAT_IN * GAT_F1, 256), 256, 0, stream>>>(Wl1, w1l, GAT_IN * GAT_F1);
  k_f32_to_bf16<<<nblk((long long)GAT_IN * GAT_F1, 256), 256, 0, stream>>>(Wr1, w1r, GAT_IN * GAT_F1);
  {
    dim3 g((Nn + 63) / 64, GAT_F1 / 64);
    k_gemm_bf16<<<g, 128, 0, stream>>>(x_bf, w1l, xl1, Nn, GAT_F1, GAT_IN);
    k_gemm_bf16<<<g, 128, 0, stream>>>(x_bf, w1r, xr1, Nn, GAT_F1, GAT_IN);
  }

  // ---- conv1 attention softmax + aggregation ----
  k_zero_u32<<<nblk(Nn * GAT_H, 256), 256, 0, stream>>>(smax1, Nn * GAT_H);
  k_zero_u32<<<nblk(Nn * GAT_H, 256), 256, 0, stream>>>((unsigned*)den1, Nn * GAT_H);
  k_zero_u32<<<nblk((long long)Nn * GAT_F1, 256), 256, 0, stream>>>((unsigned*)out1, Nn * GAT_F1);
  {
    int tasks = E * GAT_H;
    k_edge_score<<<nblk(tasks, 8), 256, 0, stream>>>(xl1, xr1, att1, src, dst, s1, smax1, E, GAT_H, GAT_C1);
    k_edge_exp<<<nblk(tasks, 256), 256, 0, stream>>>(dst, s1, smax1, den1, E, GAT_H);
    k_edge_scatter<<<nblk(tasks, 8), 256, 0, stream>>>(xl1, src, dst, s1, den1, out1, E, GAT_H, GAT_C1);
  }
  k_bias_relu_bf16<<<nblk((long long)Nn * GAT_F1, 256), 256, 0, stream>>>(out1, b1, h_bf, Nn * GAT_F1, GAT_F1);

  // ---- conv2 projections ----
  k_f32_to_bf16<<<nblk((long long)GAT_F1 * GAT_OUT, 256), 256, 0, stream>>>(Wl2, w2l, GAT_F1 * GAT_OUT);
  k_f32_to_bf16<<<nblk((long long)GAT_F1 * GAT_OUT, 256), 256, 0, stream>>>(Wr2, w2r, GAT_F1 * GAT_OUT);
  {
    dim3 g((Nn + 63) / 64, GAT_OUT / 64);
    k_gemm_bf16<<<g, 128, 0, stream>>>(h_bf, w2l, hl2, Nn, GAT_OUT, GAT_F1);
    k_gemm_bf16<<<g, 128, 0, stream>>>(h_bf, w2r, hr2, Nn, GAT_OUT, GAT_F1);
  }

  // ---- conv2 attention softmax + aggregation (directly into d_out, pre-seeded with bias) ----
  k_zero_u32<<<nblk(Nn, 256), 256, 0, stream>>>(smax2, Nn);
  k_zero_u32<<<nblk(Nn, 256), 256, 0, stream>>>((unsigned*)den2, Nn);
  k_init_bias<<<nblk((long long)Nn * GAT_OUT, 256), 256, 0, stream>>>(out, b2, Nn * GAT_OUT, GAT_OUT);
  k_edge_score<<<nblk(E, 8), 256, 0, stream>>>(hl2, hr2, att2, src, dst, s2, smax2, E, 1, GAT_OUT);
  k_edge_exp<<<nblk(E, 256), 256, 0, stream>>>(dst, s2, smax2, den2, E, 1);
  k_edge_scatter<<<nblk(E, 8), 256, 0, stream>>>(hl2, src, dst, s2, den2, out, E, 1, GAT_OUT);
}
